// LinearMultiheadAttention_86457691669061
// MI455X (gfx1250) — matrix operand order repaired and checked by reading, not compiled, not run on hardware
//
#include <hip/hip_runtime.h>
#include <hip/hip_bf16.h>
#include <cstdint>

typedef _Float16 half_t;
typedef __attribute__((ext_vector_type(16))) _Float16 v16h;
typedef __attribute__((ext_vector_type(8)))  _Float16 v8h;
typedef __attribute__((ext_vector_type(8)))  float    v8f;

// ---------------------------------------------------------------------------
// gfx1250 async global->LDS copy (ASYNCcnt-tracked, bypasses VGPRs).
// LDS operand is the wave-relative LDS byte offset (low 32 bits of the
// generic shared-memory address, per ISA aperture mapping).
// ---------------------------------------------------------------------------
__device__ __forceinline__ void async_copy_b128(void* lds_dst, const void* gsrc)
{
    uint32_t loff = (uint32_t)(uintptr_t)lds_dst;
    uint64_t gaddr = (uint64_t)(uintptr_t)gsrc;
    asm volatile("global_load_async_to_lds_b128 %0, %1, off"
                 :: "v"(loff), "v"(gaddr)
                 : "memory");
}
__device__ __forceinline__ void wait_async_all()
{
    asm volatile("s_wait_asynccnt 0x0" ::: "memory");
}

// ---------------------------------------------------------------------------
// One-time f32 -> f16 conversion (n must be a multiple of 8; true here).
// ---------------------------------------------------------------------------
__global__ void cvt_f32_f16_kernel(const float* __restrict__ src,
                                   half_t* __restrict__ dst, long n)
{
    long i = ((long)blockIdx.x * blockDim.x + threadIdx.x) * 8;
    if (i + 7 >= n + 1 && i >= n) return;
    if (i + 7 < n) {
        const float4* s4 = (const float4*)(src + i);
        float4 a = s4[0], b = s4[1];
        v8h h;
        h[0] = (half_t)a.x; h[1] = (half_t)a.y; h[2] = (half_t)a.z; h[3] = (half_t)a.w;
        h[4] = (half_t)b.x; h[5] = (half_t)b.y; h[6] = (half_t)b.z; h[7] = (half_t)b.w;
        *(v8h*)(dst + i) = h;
    }
}

// ---------------------------------------------------------------------------
// Generic tiled WMMA GEMM on f16 operands, f32 accumulate:
//   C = (A @ B + biasRow + biasCol*rowScale) * scale
// A: f16 [M,K] (or [K,M] if TRANS_A, indexed A[k*lda+m]), B: f16 [K,N].
// Block = 128 threads (4 waves). Tile 64x64, k-step 32.
// Non-transposed staging uses async global->LDS b128 copies.
// Requires M%64==0, N%64==0, K%32==0 (true for all shapes here).
// ---------------------------------------------------------------------------
template<bool TRANS_A, typename TOUT>
__global__ __launch_bounds__(128)
void gemm_wmma_kernel(const half_t* __restrict__ A, const half_t* __restrict__ B,
                      TOUT* __restrict__ C,
                      int M, int N, int K, int lda, int ldb, int ldc,
                      long sA, long sB, long sC,
                      const float* __restrict__ biasRow,
                      const float* __restrict__ biasCol,
                      const float* __restrict__ rowScale,
                      float scale)
{
    const int bz = blockIdx.z;
    A += (long)bz * sA;  B += (long)bz * sB;  C += (long)bz * sC;

    const int n0   = blockIdx.x * 64;
    const int m0   = blockIdx.y * 64;
    const int tid  = threadIdx.x;
    const int lane = tid & 31;
    const int wave = tid >> 5;

    __shared__ half_t As[64][40];   // row stride 80B (16B-aligned), 32 used cols
    __shared__ half_t Bs[32][72];   // row stride 144B (16B-aligned), 64 used cols

    const int mw = wave * 16;
    v8f acc[4] = {};

    for (int k0 = 0; k0 < K; k0 += 32) {
        // ---- stage A tile ----
        if (TRANS_A) {
            // transpose during copy: scalar ds stores (not b128-contiguous)
            for (int i = tid; i < 64 * 32; i += 128) {
                int c = i >> 6;          // k within step
                int r = i & 63;          // m within tile
                As[r][c] = A[(long)(k0 + c) * lda + (m0 + r)];
            }
        } else {
            // 64 rows x 4 chunks of 16B, async copy
            for (int i = tid; i < 256; i += 128) {
                int r = i >> 2, c = (i & 3) * 8;
                async_copy_b128(&As[r][c], A + (long)(m0 + r) * lda + k0 + c);
            }
        }
        // ---- stage B tile: 32 rows x 8 chunks of 16B, async copy ----
        for (int i = tid; i < 256; i += 128) {
            int r = i >> 3, c = (i & 7) * 8;
            async_copy_b128(&Bs[r][c], B + (long)(k0 + r) * ldb + n0 + c);
        }
        wait_async_all();
        __syncthreads();

        // ---- A fragment (16x32 f16, CDNA5 layout) ----
        const int arow  = mw + (lane & 15);
        const int koffA = (lane >> 4) * 8;       // lanes 16-31 hold K+8
        v16h af;
        #pragma unroll
        for (int j = 0; j < 8; ++j) {
            int kk = ((j < 4) ? (2 * j) : (16 + 2 * (j - 4))) + koffA;
            af[2 * j]     = As[arow][kk];
            af[2 * j + 1] = As[arow][kk + 1];
        }

        const int bcol  = lane & 15;
        const int koffB = (lane >> 4) * 8;       // lanes 16-31 hold K+8
        #pragma unroll
        for (int nt = 0; nt < 4; ++nt) {
            v16h bf;
            #pragma unroll
            for (int hh = 0; hh < 16; ++hh)
                bf[hh] = Bs[koffB + (hh & 7) + ((hh >> 3) << 4)][nt * 16 + bcol];
            acc[nt] = __builtin_amdgcn_wmma_f32_16x16x32_f16(
                false, af, false, bf, (short)0, acc[nt], false, false);
        }
        __syncthreads();
    }

    // ---- epilogue: C layout: lanes 0-15 -> M=v, lanes 16-31 -> M=8+v ----
    const int crow0 = m0 + mw + ((lane >> 4) << 3);
    const int ccol0 = n0 + (lane & 15);
    #pragma unroll
    for (int nt = 0; nt < 4; ++nt) {
        int col  = ccol0 + nt * 16;
        float cb = biasCol ? biasCol[col] : 0.0f;
        #pragma unroll
        for (int v = 0; v < 8; ++v) {
            int row  = crow0 + v;
            float rb = biasRow  ? biasRow[row]  : 0.0f;
            float rs = rowScale ? rowScale[row] : 1.0f;
            float val = (acc[nt][v] + rb + cb * rs) * scale;
            C[(long)row * ldc + col] = (TOUT)val;
        }
    }
}

// ---------------------------------------------------------------------------
// colsum[j] = sum_i W[i*PKdim + j]   (exact bias folding: We^T @ bk term)
// ---------------------------------------------------------------------------
__global__ void colsum_kernel(const float* __restrict__ W, float* __restrict__ out,
                              int Sdim, int PKdim)
{
    int j = blockIdx.x * blockDim.x + threadIdx.x;
    if (j >= PKdim) return;
    float s = 0.0f;
    for (int i = 0; i < Sdim; ++i) s += W[(long)i * PKdim + j];
    out[j] = s;
}

// ---------------------------------------------------------------------------
// Fused attention per (b, h, 64 query rows):
//   async LDS <- Kp/Vp head slice [256x64 f16] + q rows [64x64 f16]
//   scores = q @ Kp^T (WMMA) -> LDS f32, per-row softmax -> in-place f16 probs
//   out    = probs @ Vp (WMMA) -> f16 global [B,S,H*DV]
// Block = 128 threads (4 waves), wave w owns query rows w*16..w*16+15.
// ---------------------------------------------------------------------------
__global__ __launch_bounds__(128)
void attn_kernel(const half_t* __restrict__ q,    // [B,S,H*DK], pre-scaled
                 const half_t* __restrict__ Kp,   // [B,PK,H*DK]
                 const half_t* __restrict__ Vp,   // [B,PK,H*DV]
                 half_t* __restrict__ out)        // [B,S,H*DV] f16
{
    constexpr int S = 8192, H = 8, DK = 64, PK = 256, HD = H * DK;
    const int sblk = blockIdx.x;
    const int h    = blockIdx.y;
    const int b    = blockIdx.z;
    const int tid  = threadIdx.x;
    const int lane = tid & 31;
    const int wave = tid >> 5;

    extern __shared__ char smem[];
    half_t (*Ks)[72] = (half_t(*)[72])(smem);                       // 256x72 halves
    half_t (*Vs)[72] = (half_t(*)[72])(smem + 256 * 72 * 2);
    half_t (*qs)[72] = (half_t(*)[72])(smem + 2 * 256 * 72 * 2);    // 64x72
    float*  sc       = (float*)(smem + 2 * 256 * 72 * 2 + 64 * 72 * 2); // 4x16x264 f32

    const long kbase = ((long)b * PK) * HD + (long)h * DK;
    const long qbase = ((long)b * S + (long)sblk * 64) * HD + (long)h * DK;

    // ---- async stage K, V (256 rows x 8 b128 chunks) and q (64 x 8) ----
    for (int i = tid; i < 2048; i += 128) {
        int p = i >> 3, c = (i & 7) * 8;
        async_copy_b128(&Ks[p][c], Kp + kbase + (long)p * HD + c);
        async_copy_b128(&Vs[p][c], Vp + kbase + (long)p * HD + c);
    }
    for (int i = tid; i < 512; i += 128) {
        int r = i >> 3, c = (i & 7) * 8;
        async_copy_b128(&qs[r][c], q + qbase + (long)r * HD + c);
    }
    wait_async_all();
    __syncthreads();

    float* scw = sc + wave * 16 * 264;

    // ---- scores = q @ Kp^T : 16 col-tiles x 2 k-steps of 32 ----
    {
        const int arow  = wave * 16 + (lane & 15);
        const int koffA = (lane >> 4) * 8;
        v16h a0, a1;
        #pragma unroll
        for (int j = 0; j < 8; ++j) {
            int kk = ((j < 4) ? (2 * j) : (16 + 2 * (j - 4))) + koffA;
            a0[2 * j]     = qs[arow][kk];
            a0[2 * j + 1] = qs[arow][kk + 1];
            a1[2 * j]     = qs[arow][kk + 32];
            a1[2 * j + 1] = qs[arow][kk + 33];
        }
        const int bcol  = lane & 15;
        const int koffB = (lane >> 4) * 8;
        const int r0    = (lane >> 4) << 3;
        for (int pt = 0; pt < 16; ++pt) {
            v16h b0, b1;                         // B(d,p)=Kp[p][d]: contiguous in d
            const half_t* krow = &Ks[pt * 16 + bcol][0];
            #pragma unroll
            for (int hh = 0; hh < 16; ++hh) {
                b0[hh] = krow[koffB + (hh & 7) + ((hh >> 3) << 4)];
                b1[hh] = krow[32 + koffB + (hh & 7) + ((hh >> 3) << 4)];
            }
            v8f accs = {};
            accs = __builtin_amdgcn_wmma_f32_16x16x32_f16(false, a0, false, b0,
                                                          (short)0, accs, false, false);
            accs = __builtin_amdgcn_wmma_f32_16x16x32_f16(false, a1, false, b1,
                                                          (short)0, accs, false, false);
            int col = pt * 16 + (lane & 15);
            #pragma unroll
            for (int v = 0; v < 8; ++v)
                scw[(r0 + v) * 264 + col] = accs[v];
        }
    }
    __syncthreads();

    // ---- per-row softmax over PK=256 (threads 0..63, one row each) ----
    if (tid < 64) {
        float* row = sc + (tid >> 4) * 16 * 264 + (tid & 15) * 264;
        float mx = -1e30f;
        for (int j = 0; j < 256; ++j) mx = fmaxf(mx, row[j]);
        float sum = 0.0f;
        half_t* prow = (half_t*)row;     // in-place f32 -> f16 (write idx < read idx)
        for (int j = 0; j < 256; ++j) {
            float e = __expf(row[j] - mx);
            sum += e;
            prow[j] = (half_t)e;
        }
        float inv = 1.0f / sum;
        for (int j = 0; j < 256; ++j)
            prow[j] = (half_t)((float)prow[j] * inv);
    }
    __syncthreads();

    // ---- out = probs @ Vp : K=256 -> 8 k-steps, 4 col-tiles of 16 ----
    {
        half_t* pbase   = (half_t*)scw;      // probs rows, stride 528 halves
        const int arow  = lane & 15;
        const int koffA = (lane >> 4) * 8;
        const int bcol  = lane & 15;
        const int koffB = (lane >> 4) * 8;
        v8f o[4] = {};
        for (int kt = 0; kt < 8; ++kt) {
            v16h af;
            const half_t* pr = pbase + arow * 528;
            #pragma unroll
            for (int j = 0; j < 8; ++j) {
                int kk = kt * 32 + ((j < 4) ? (2 * j) : (16 + 2 * (j - 4))) + koffA;
                af[2 * j]     = pr[kk];
                af[2 * j + 1] = pr[kk + 1];
            }
            #pragma unroll
            for (int nt = 0; nt < 4; ++nt) {
                v16h bf;
                #pragma unroll
                for (int hh = 0; hh < 16; ++hh)
                    bf[hh] = Vs[kt * 32 + koffB + (hh & 7) + ((hh >> 3) << 4)][nt * 16 + bcol];
                o[nt] = __builtin_amdgcn_wmma_f32_16x16x32_f16(
                    false, af, false, bf, (short)0, o[nt], false, false);
            }
        }
        const long obase = ((long)b * S + (long)sblk * 64 + wave * 16) * HD + (long)h * DK;
        const int r0  = (lane >> 4) << 3;
        const int col = lane & 15;
        #pragma unroll
        for (int nt = 0; nt < 4; ++nt)
            #pragma unroll
            for (int v = 0; v < 8; ++v)
                out[obase + (long)(r0 + v) * HD + nt * 16 + col] = (half_t)o[nt][v];
    }
}

// ---------------------------------------------------------------------------
extern "C" void kernel_launch(void* const* d_in, const int* in_sizes, int n_in,
                              void* d_out, int out_size, void* d_ws, size_t ws_size,
                              hipStream_t stream)
{
    (void)in_sizes; (void)n_in; (void)out_size; (void)ws_size;
    constexpr int B = 4, S = 8192, D = 512, H = 8, DK = 64, PK = 256;
    constexpr int HD = H * DK;           // 512
    constexpr long BS = (long)B * S;     // 32768

    const float* query = (const float*)d_in[0];
    const float* value = (const float*)d_in[1];
    const float* Wq = (const float*)d_in[2];
    const float* bq = (const float*)d_in[3];
    const float* Wk = (const float*)d_in[4];
    const float* bk = (const float*)d_in[5];
    const float* Wv = (const float*)d_in[6];
    const float* bv = (const float*)d_in[7];
    const float* We = (const float*)d_in[8];
    const float* be = (const float*)d_in[9];
    const float* Wf = (const float*)d_in[10];
    const float* bf = (const float*)d_in[11];
    const float* Wo = (const float*)d_in[12];
    const float* bo = (const float*)d_in[13];
    float* out = (float*)d_out;

    // ---- workspace layout ----
    char* ws = (char*)d_ws;
    auto take = [&](long bytes) { char* p = ws; ws += (bytes + 255) & ~255L; return p; };
    half_t* query16 = (half_t*)take(BS * D * 2);            // 33.5 MB
    half_t* value16 = (half_t*)take(BS * D * 2);            // 33.5 MB
    half_t* Wq16    = (half_t*)take((long)D * HD * 2);
    half_t* Wk16    = (half_t*)take((long)D * HD * 2);
    half_t* Wv16    = (half_t*)take((long)D * HD * 2);
    half_t* Wo16    = (half_t*)take((long)HD * D * 2);
    half_t* We16    = (half_t*)take((long)S * PK * 2);
    half_t* Wf16    = (half_t*)take((long)S * PK * 2);
    half_t* qf16    = (half_t*)take(BS * HD * 2);           // 33.5 MB
    half_t* T       = (half_t*)take((long)B * PK * D * 2);
    half_t* U       = (half_t*)take((long)B * PK * D * 2);
    half_t* Kp      = (half_t*)take((long)B * PK * HD * 2);
    half_t* Vp      = (half_t*)take((long)B * PK * HD * 2);
    half_t* attn16  = (half_t*)take(BS * HD * 2);           // 33.5 MB
    float*  csWe    = (float*)take(PK * sizeof(float));
    float*  csWf    = (float*)take(PK * sizeof(float));

    // ---- 0) one-time f32 -> f16 conversions ----
    auto cvt = [&](const float* s, half_t* d, long n) {
        cvt_f32_f16_kernel<<<dim3((unsigned)(n / (256 * 8))), 256, 0, stream>>>(s, d, n);
    };
    cvt(query, query16, BS * D);
    cvt(value, value16, BS * D);
    cvt(Wq, Wq16, (long)D * HD);
    cvt(Wk, Wk16, (long)D * HD);
    cvt(Wv, Wv16, (long)D * HD);
    cvt(Wo, Wo16, (long)HD * D);
    cvt(We, We16, (long)S * PK);
    cvt(Wf, Wf16, (long)S * PK);

    // ---- 1) column sums of We, Wf (bias folding) ----
    colsum_kernel<<<1, 256, 0, stream>>>(We, csWe, S, PK);
    colsum_kernel<<<1, 256, 0, stream>>>(Wf, csWf, S, PK);

    // ---- 2) q = (query @ Wq + bq) * DK^-0.5 -> f16 [B*S, 512] ----
    gemm_wmma_kernel<false, half_t><<<dim3(HD / 64, BS / 64, 1), 128, 0, stream>>>(
        query16, Wq16, qf16, (int)BS, HD, D, D, HD, HD, 0, 0, 0,
        nullptr, bq, nullptr, 0.125f);

    // ---- 3) T[b] = We^T @ value[b], U[b] = Wf^T @ value[b]  [PK, D] ----
    gemm_wmma_kernel<true, half_t><<<dim3(D / 64, PK / 64, B), 128, 0, stream>>>(
        We16, value16, T, PK, D, S, PK, D, D,
        0, (long)S * D, (long)PK * D, nullptr, nullptr, nullptr, 1.0f);
    gemm_wmma_kernel<true, half_t><<<dim3(D / 64, PK / 64, B), 128, 0, stream>>>(
        Wf16, value16, U, PK, D, S, PK, D, D,
        0, (long)S * D, (long)PK * D, nullptr, nullptr, nullptr, 1.0f);

    // ---- 4) Kp = T@Wk + be[:,None] + colsum(We)[:,None]*bk ;  Vp likewise ----
    gemm_wmma_kernel<false, half_t><<<dim3(HD / 64, PK / 64, B), 128, 0, stream>>>(
        T, Wk16, Kp, PK, HD, D, D, HD, HD,
        (long)PK * D, 0, (long)PK * HD, be, bk, csWe, 1.0f);
    gemm_wmma_kernel<false, half_t><<<dim3(HD / 64, PK / 64, B), 128, 0, stream>>>(
        U, Wv16, Vp, PK, HD, D, D, HD, HD,
        (long)PK * D, 0, (long)PK * HD, bf, bv, csWf, 1.0f);

    // ---- 5) fused attention ----
    constexpr size_t ATTN_SMEM =
        2 * 256 * 72 * sizeof(half_t) + 64 * 72 * sizeof(half_t) +
        4 * 16 * 264 * sizeof(float);   // 150528 bytes, < 320 KB WGP LDS
    attn_kernel<<<dim3(S / 64, H, B), 128, ATTN_SMEM, stream>>>(qf16, Kp, Vp, attn16);

    // ---- 6) out = attn @ Wo + bo -> f32 d_out ----
    gemm_wmma_kernel<false, float><<<dim3(D / 64, BS / 64, 1), 128, 0, stream>>>(
        attn16, Wo16, out, (int)BS, D, HD, HD, D, D, 0, 0, 0,
        nullptr, bo, nullptr, 1.0f);
}
